// MultiHeadAttention_15341623181946
// MI455X (gfx1250) — matrix operand order repaired and checked by reading, not compiled, not run on hardware
//
#include <hip/hip_runtime.h>

typedef __attribute__((ext_vector_type(16))) _Float16 v16h;
typedef __attribute__((ext_vector_type(8)))  float    v8f;
typedef __attribute__((ext_vector_type(4)))  _Float16 v4h;

#define SEQ    1024
#define NHEAD  16
#define DHEAD  64
#define DMODEL 1024

// ---------------------------------------------------------------------------
// CDNA5 async copy: 16 bytes global -> LDS per enabled lane, tracked by
// ASYNCcnt (no VGPR round trip). LDS address = low 32 bits of the generic
// pointer (flat->LDS truncation, ISA 10.2). Must s_wait_asynccnt before any
// barrier/read since the compiler's DScnt bookkeeping doesn't see the write.
// ---------------------------------------------------------------------------
__device__ __forceinline__ void async_copy16(void* lds_dst, const void* gsrc) {
  uint32_t loff = (uint32_t)(uintptr_t)lds_dst;
  unsigned long long ga = (unsigned long long)(uintptr_t)gsrc;
  asm volatile("global_load_async_to_lds_b128 %0, %1, off"
               :: "v"(loff), "v"(ga)
               : "memory");
}
__device__ __forceinline__ void wait_async() {
  asm volatile("s_wait_asynccnt 0" ::: "memory");
}

// ---------------------------------------------------------------------------
// Cooperative staging of a 64x32 tile into LDS (f16, dst row stride 48).
// f32 source: VALU convert path (async copy cannot convert).
// f16 source: async global->LDS path.
// 128-thread blocks.
// ---------------------------------------------------------------------------
__device__ __forceinline__ void stage64x32(_Float16* dst, const float* src, long ld) {
  for (int i = threadIdx.x; i < 64 * 8; i += 128) {
    int r = i >> 3, c = (i & 7) << 2;
    float4 t = *reinterpret_cast<const float4*>(src + (long)r * ld + c);
    v4h h;
    h.x = (_Float16)t.x; h.y = (_Float16)t.y;
    h.z = (_Float16)t.z; h.w = (_Float16)t.w;
    *reinterpret_cast<v4h*>(dst + r * 48 + c) = h;
  }
}

__device__ __forceinline__ void stage64x32(_Float16* dst, const _Float16* src, long ld) {
  for (int i = threadIdx.x; i < 64 * 4; i += 128) {
    int r = i >> 2, c = (i & 3) << 3;
    async_copy16(dst + r * 48 + c, src + (long)r * ld + c);
  }
}

// Fence matching the staging path used for type T
__device__ __forceinline__ void wait_stage(const float*) {}
__device__ __forceinline__ void wait_stage(const _Float16*) { wait_async(); }

// ---------------------------------------------------------------------------
// WMMA GEMM: C[m][n] = act( (sum_k A[m][k]*W[n][k] + bias[n]) * scale )
// Block = 128 threads (4 waves), computes a 64x64 C tile; wave w owns rows
// 16w..16w+15 as four 16x16 WMMA accumulators. K stepped by 32.
// ---------------------------------------------------------------------------
template <typename AT, typename OT, bool RELU>
__global__ __launch_bounds__(128) void wmma_gemm(const AT* __restrict__ A, long lda,
                                                 const float* __restrict__ W, long ldw,
                                                 const float* __restrict__ bias,
                                                 OT* __restrict__ C, long ldc,
                                                 int K, float scale) {
  const int n0  = blockIdx.x * 64;
  const int m0  = blockIdx.y * 64;
  const int w   = threadIdx.x >> 5;
  const int ln  = threadIdx.x & 31;
  const int hl  = ln >> 4;
  const int l16 = ln & 15;

  __shared__ _Float16 as[64 * 48];
  __shared__ _Float16 bs[64 * 48];

  v8f acc[4] = {};

  for (int k0 = 0; k0 < K; k0 += 32) {
    __syncthreads();
    stage64x32(as, A + (long)m0 * lda + k0, lda);
    stage64x32(bs, W + (long)n0 * ldw + k0, ldw);
    wait_stage(A);
    __syncthreads();

    // A fragment (16x32) for this wave's row strip
    v16h a;
#pragma unroll
    for (int j = 0; j < 16; ++j) {
      int kk = 16 * hl + j;
      a[j] = as[(16 * w + l16) * 48 + kk];
    }
#pragma unroll
    for (int nt = 0; nt < 4; ++nt) {
      v16h b;  // B[k][n] = W[n0+nt*16+n][k0+k]
#pragma unroll
      for (int j = 0; j < 16; ++j)
        b[j] = bs[(nt * 16 + l16) * 48 + 16 * hl + j];
      acc[nt] = __builtin_amdgcn_wmma_f32_16x16x32_f16(false, a, false, b,
                                                       (short)0, acc[nt], false, false);
    }
  }

#pragma unroll
  for (int nt = 0; nt < 4; ++nt) {
    int n = n0 + nt * 16 + l16;
    float bb = bias[n];
#pragma unroll
    for (int r = 0; r < 8; ++r) {
      int m = m0 + 16 * w + r + 8 * hl;
      float vv = (acc[nt][r] + bb) * scale;
      if (RELU) vv = fmaxf(vv, 0.0f);
      C[(long)m * ldc + n] = (OT)vv;
    }
  }
}

// ---------------------------------------------------------------------------
// Flash attention. grid = (16 qtiles, 16 heads, 8 batch), block = 128 = 4 waves.
// Each wave: one 16-row Q tile; block shares K/V 32x64 LDS tiles per key step,
// staged with async global->LDS copies. qh is pre-scaled by 1/sqrt(d).
// qh/kh/vh are f16 [8192][1024] with head h at column h*64.
// ---------------------------------------------------------------------------
__global__ __launch_bounds__(128) void flash_attn(const _Float16* __restrict__ qh,
                                                  const _Float16* __restrict__ kh,
                                                  const _Float16* __restrict__ vh,
                                                  _Float16* __restrict__ y) {
  const int b   = blockIdx.z;
  const int h   = blockIdx.y;
  const int q0  = blockIdx.x * 64;
  const int w   = threadIdx.x >> 5;
  const int ln  = threadIdx.x & 31;
  const int hl  = ln >> 4;
  const int l16 = ln & 15;

  __shared__ _Float16 qs[64 * 72];
  __shared__ _Float16 ks[32 * 72];
  __shared__ _Float16 vs[32 * 72];
  __shared__ _Float16 ps[4][16 * 40];

  const long rs_g = NHEAD * DHEAD;  // 1024
  const _Float16* qbase = qh + ((long)(b * SEQ + q0)) * rs_g + h * DHEAD;
  const _Float16* kbase = kh + ((long)(b * SEQ)) * rs_g + h * DHEAD;
  const _Float16* vbase = vh + ((long)(b * SEQ)) * rs_g + h * DHEAD;

  // stage Q 64x64 via async copies
  for (int i = threadIdx.x; i < 64 * 8; i += 128) {
    int r = i >> 3, c = (i & 7) << 3;
    async_copy16(&qs[r * 72 + c], &qbase[(long)r * rs_g + c]);
  }
  wait_async();
  __syncthreads();

  // Q fragments (dims 0-31 and 32-63) for this wave
  v16h aq0, aq1;
#pragma unroll
  for (int j = 0; j < 16; ++j) {
    int kk = 16 * hl + j;
    aq0[j] = qs[(16 * w + l16) * 72 + kk];
    aq1[j] = qs[(16 * w + l16) * 72 + 32 + kk];
  }

  float rmax[8], rsum[8];
#pragma unroll
  for (int r = 0; r < 8; ++r) { rmax[r] = -1e30f; rsum[r] = 0.0f; }
  v8f o0 = {}, o1 = {}, o2 = {}, o3 = {};

  for (int kt = 0; kt < SEQ / 32; ++kt) {
    if (kt + 1 < SEQ / 32) {  // prefetch next K/V tiles (global_prefetch_b8)
      __builtin_prefetch(&kbase[(long)((kt + 1) * 32 + ln) * rs_g], 0, 3);
      __builtin_prefetch(&vbase[(long)((kt + 1) * 32 + ln) * rs_g], 0, 3);
    }
    __syncthreads();  // previous iteration's readers done
    for (int i = threadIdx.x; i < 32 * 8; i += 128) {
      int r = i >> 3, c = (i & 7) << 3;
      async_copy16(&ks[r * 72 + c], &kbase[(long)(kt * 32 + r) * rs_g + c]);
      async_copy16(&vs[r * 72 + c], &vbase[(long)(kt * 32 + r) * rs_g + c]);
    }
    wait_async();
    __syncthreads();

    // S = Q K^T : two 16x16 score tiles (keys 0-15 and 16-31 of this step)
    v8f s0 = {}, s1 = {};
    {
      v16h bk;
#pragma unroll
      for (int j = 0; j < 16; ++j) bk[j] = ks[l16 * 72 + 16 * hl + j];
      s0 = __builtin_amdgcn_wmma_f32_16x16x32_f16(false, aq0, false, bk, (short)0, s0, false, false);
#pragma unroll
      for (int j = 0; j < 16; ++j) bk[j] = ks[l16 * 72 + 32 + 16 * hl + j];
      s0 = __builtin_amdgcn_wmma_f32_16x16x32_f16(false, aq1, false, bk, (short)0, s0, false, false);
#pragma unroll
      for (int j = 0; j < 16; ++j) bk[j] = ks[(16 + l16) * 72 + 16 * hl + j];
      s1 = __builtin_amdgcn_wmma_f32_16x16x32_f16(false, aq0, false, bk, (short)0, s1, false, false);
#pragma unroll
      for (int j = 0; j < 16; ++j) bk[j] = ks[(16 + l16) * 72 + 32 + 16 * hl + j];
      s1 = __builtin_amdgcn_wmma_f32_16x16x32_f16(false, aq1, false, bk, (short)0, s1, false, false);
    }

    // Online softmax. C layout: VGPR r <-> row r + 8*hl, col = l16 (+16 for s1).
    // Row statistics reduce across the 16 lanes of each half (xor masks < 16).
#pragma unroll
    for (int r = 0; r < 8; ++r) {
      float mx = fmaxf(s0[r], s1[r]);
#pragma unroll
      for (int d = 1; d < 16; d <<= 1) mx = fmaxf(mx, __shfl_xor(mx, d, 32));
      float mnew = fmaxf(rmax[r], mx);
      float cf = __expf(rmax[r] - mnew);
      float p0 = __expf(s0[r] - mnew);
      float p1 = __expf(s1[r] - mnew);
      float pt = p0 + p1;
#pragma unroll
      for (int d = 1; d < 16; d <<= 1) pt += __shfl_xor(pt, d, 32);
      rsum[r] = rsum[r] * cf + pt;
      rmax[r] = mnew;
      o0[r] *= cf; o1[r] *= cf; o2[r] *= cf; o3[r] *= cf;
      ps[w][(r + 8 * hl) * 40 + l16]      = (_Float16)p0;
      ps[w][(r + 8 * hl) * 40 + 16 + l16] = (_Float16)p1;
    }

    // Re-layout probs C-fragment -> A-fragment through LDS (same-wave DS order)
    v16h ap;
#pragma unroll
    for (int j = 0; j < 16; ++j) {
      int kk = 16 * hl + j;
      ap[j] = ps[w][l16 * 40 + kk];
    }

    // O += P V  (four 16x16 output tiles across the 64-wide head dim)
    {
      v16h bv;
#pragma unroll
      for (int j = 0; j < 16; ++j) bv[j] = vs[(16 * hl + j) * 72 + l16];
      o0 = __builtin_amdgcn_wmma_f32_16x16x32_f16(false, ap, false, bv, (short)0, o0, false, false);
#pragma unroll
      for (int j = 0; j < 16; ++j) bv[j] = vs[(16 * hl + j) * 72 + 16 + l16];
      o1 = __builtin_amdgcn_wmma_f32_16x16x32_f16(false, ap, false, bv, (short)0, o1, false, false);
#pragma unroll
      for (int j = 0; j < 16; ++j) bv[j] = vs[(16 * hl + j) * 72 + 32 + l16];
      o2 = __builtin_amdgcn_wmma_f32_16x16x32_f16(false, ap, false, bv, (short)0, o2, false, false);
#pragma unroll
      for (int j = 0; j < 16; ++j) bv[j] = vs[(16 * hl + j) * 72 + 48 + l16];
      o3 = __builtin_amdgcn_wmma_f32_16x16x32_f16(false, ap, false, bv, (short)0, o3, false, false);
    }
  }

  // Normalize and store y[b, q, h*64 + d] as f16
  _Float16* ybase = y + ((long)(b * SEQ + q0 + 16 * w)) * rs_g + h * DHEAD;
#pragma unroll
  for (int r = 0; r < 8; ++r) {
    float inv = 1.0f / rsum[r];
    int row = r + 8 * hl;
    ybase[(long)row * rs_g + l16]      = (_Float16)(o0[r] * inv);
    ybase[(long)row * rs_g + 16 + l16] = (_Float16)(o1[r] * inv);
    ybase[(long)row * rs_g + 32 + l16] = (_Float16)(o2[r] * inv);
    ybase[(long)row * rs_g + 48 + l16] = (_Float16)(o3[r] * inv);
  }
}

// ---------------------------------------------------------------------------
extern "C" void kernel_launch(void* const* d_in, const int* in_sizes, int n_in,
                              void* d_out, int out_size, void* d_ws, size_t ws_size,
                              hipStream_t stream) {
  (void)in_sizes; (void)n_in; (void)out_size; (void)ws_size;

  const float* q  = (const float*)d_in[0];
  const float* k  = (const float*)d_in[1];
  const float* v  = (const float*)d_in[2];
  const float* Wq = (const float*)d_in[3];
  const float* bq = (const float*)d_in[4];
  const float* W1 = (const float*)d_in[5];
  const float* b1 = (const float*)d_in[6];
  const float* W2 = (const float*)d_in[7];
  const float* b2 = (const float*)d_in[8];
  float* out = (float*)d_out;

  const size_t SZ = (size_t)8 * SEQ * DMODEL;  // 8M elements per intermediate
  _Float16* qh = (_Float16*)d_ws;
  _Float16* kh = qh + SZ;
  _Float16* vh = kh + SZ;
  _Float16* y  = vh + SZ;
  _Float16* hm = y + SZ;

  dim3 blk(128);
  // Projections: M=8192, N=1024, K=64. Q gets the 1/sqrt(64) scale folded in.
  dim3 gproj(DMODEL / 64, (8 * SEQ) / 64);
  wmma_gemm<float, _Float16, false><<<gproj, blk, 0, stream>>>(q, 64, Wq, 64, bq, qh, DMODEL, 64, 0.125f);
  wmma_gemm<float, _Float16, false><<<gproj, blk, 0, stream>>>(k, 64, Wq, 64, bq, kh, DMODEL, 64, 1.0f);
  wmma_gemm<float, _Float16, false><<<gproj, blk, 0, stream>>>(v, 64, Wq, 64, bq, vh, DMODEL, 64, 1.0f);

  // Attention: 16 q-tiles x 16 heads x 8 batch
  dim3 gattn(SEQ / 64, NHEAD, 8);
  flash_attn<<<gattn, blk, 0, stream>>>(qh, kh, vh, y);

  // MLP: M=8192, N=1024, K=1024 (ReLU fused into layer 1)
  dim3 gmlp(DMODEL / 64, (8 * SEQ) / 64);
  wmma_gemm<_Float16, _Float16, true><<<gmlp, blk, 0, stream>>>(y, DMODEL, W1, DMODEL, b1, hm, DMODEL, DMODEL, 1.0f);
  wmma_gemm<_Float16, float, false><<<gmlp, blk, 0, stream>>>(hm, DMODEL, W2, DMODEL, b2, out, DMODEL, DMODEL, 1.0f);
}
